// MolGCN_4518305595465
// MI455X (gfx1250) — compile-verified, matrix operand order checked by reading, not run on hardware
//
#include <hip/hip_runtime.h>
#include <hip/hip_bf16.h>
#include <math.h>

// ---------------------------------------------------------------------------
// MolGCN forward for MI455X (gfx1250, wave32, WMMA).
// Heavy work = batched 128x128x128 matmuls (adj @ edge_data, adj @ adj0) done
// with V_WMMA_F32_16X16X32_F16 (A = 0/1 adjacency, exact in f16; B = tanh of
// 0.1-scale embeddings, f16-safe; f32 accumulate). Everything else collapses
// to cheap elementwise/reduction kernels because adj entries are exactly 0/1.
// ---------------------------------------------------------------------------

#define BB 64
#define NN 128
#define NAL 40
#define EED 5
#define NGC1 50
#define NCLASS 12

typedef __attribute__((ext_vector_type(16))) _Float16 v16h;
typedef __attribute__((ext_vector_type(8)))  float    v8f;

// ---------------------------------------------------------------------------
// Setup: edge embedding (tanh * adj), f16 copies (A row-major, B col-major),
// adj0, adj_cur, adj_sum = 0.
// ---------------------------------------------------------------------------
__global__ void k_setup_edge(const float* __restrict__ adjs,
                             const int* __restrict__ bfts,
                             const float* __restrict__ edge_emb,
                             float* __restrict__ edge_f32,      // [B][5][N][N]
                             _Float16* __restrict__ edge_h_cm,  // [B][5][N][N] col-major
                             float* __restrict__ adj0,          // [B][N][N]
                             _Float16* __restrict__ adj0_h_cm,  // col-major
                             float* __restrict__ adj_cur,       // [B][N][N]
                             _Float16* __restrict__ adj_cur_h,  // row-major (A operand)
                             float* __restrict__ adj_sum)
{
  int t = blockIdx.x * blockDim.x + threadIdx.x;   // b*N*N + i*N + j
  int b = t >> 14;
  int rem = t & 16383;
  int i = rem >> 7, j = rem & 127;
  float a = adjs[t];
  int e = bfts[t];
#pragma unroll
  for (int c = 0; c < EED; ++c) {
    float v = tanhf(edge_emb[e * EED + c]) * a;
    size_t base = ((size_t)(b * EED + c)) << 14;
    edge_f32[base + (i << 7) + j] = v;
    edge_h_cm[base + (j << 7) + i] = (_Float16)v;  // col-major: [col j][row i]
  }
  float a0 = fmaxf(a - (i == j ? 1.0f : 0.0f), 0.0f);
  adj0[t] = a0;
  adj0_h_cm[((size_t)b << 14) + (j << 7) + i] = (_Float16)a0;
  adj_cur[t] = a0;
  adj_cur_h[t] = (_Float16)a0;
  adj_sum[t] = 0.0f;
}

// ---------------------------------------------------------------------------
// Setup: nz, node_data = tanh(concat(node_emb, axfms)) * nz, deg0, rowmax0.
// ---------------------------------------------------------------------------
__global__ void k_setup_node(const float* __restrict__ adjs,
                             const int* __restrict__ afms,
                             const float* __restrict__ axfms,   // [B][N][35]
                             const float* __restrict__ node_emb,
                             const float* __restrict__ adj0,
                             float* __restrict__ node_out,      // [B][40][N]
                             float* __restrict__ nz,
                             float* __restrict__ deg,
                             float* __restrict__ rowmax)
{
  int b = blockIdx.x, i = threadIdx.x;
  const float* arow = adjs + ((size_t)(b * NN + i)) * NN;
  const float* a0row = adj0 + ((size_t)(b * NN + i)) * NN;
  float mx = 0.0f, d = 0.0f;
  for (int j = 0; j < NN; ++j) { mx = fmaxf(mx, arow[j]); d += a0row[j]; }
  nz[b * NN + i] = mx;
  deg[b * NN + i] = d;
  rowmax[b * NN + i] = (d > 0.0f) ? 1.0f : 0.0f;
  int af = afms[b * NN + i];
#pragma unroll 4
  for (int c = 0; c < NAL; ++c) {
    float v = (c < EED) ? node_emb[af * EED + c]
                        : axfms[((size_t)(b * NN + i)) * 35 + (c - EED)];
    node_out[((size_t)(b * NAL + c)) * NN + i] = tanhf(v) * mx;
  }
}

// ---------------------------------------------------------------------------
// Fingerprint: fp[b,5g+o] = tanh( sum_k W_out[g,o,k] * sum_i node[4g+k,i]*nz )
// ---------------------------------------------------------------------------
__global__ void k_fp(const float* __restrict__ node_cur,
                     const float* __restrict__ nz,
                     const float* __restrict__ Wout_r,   // 10*5*4
                     float* __restrict__ fp_out)         // [B][50] (pre-offset)
{
  __shared__ float s[NAL];
  int b = blockIdx.x, t = threadIdx.x;
  if (t < NAL) {
    float acc = 0.0f;
    const float* row = node_cur + ((size_t)(b * NAL + t)) * NN;
    const float* z = nz + b * NN;
    for (int i = 0; i < NN; ++i) acc += row[i] * z[i];
    s[t] = acc;
  }
  __syncthreads();
  if (t < NGC1) {
    int g = t / 5, o = t % 5;
    float acc = 0.0f;
#pragma unroll
    for (int k = 0; k < 4; ++k) acc += Wout_r[(g * 5 + o) * 4 + k] * s[4 * g + k];
    fp_out[b * NGC1 + t] = tanhf(acc);
  }
}

// ---------------------------------------------------------------------------
// E[b,c,i] = (optional W_edge mix of) F[b,c,i] = sum_j adj[b,i,j]*X[b,c,i,j]
// ---------------------------------------------------------------------------
__global__ void k_emix(const float* __restrict__ adj,
                       const float* __restrict__ X,      // [B][5][N][N]
                       const float* __restrict__ Wmix,   // 5x5 or nullptr
                       float* __restrict__ E)            // [B][5][N]
{
  int b = blockIdx.x, i = threadIdx.x;
  float F[EED] = {0.f, 0.f, 0.f, 0.f, 0.f};
  const float* arow = adj + ((size_t)(b * NN + i)) * NN;
  for (int j = 0; j < NN; ++j) {
    float a = arow[j];
    if (a != 0.0f) {
#pragma unroll
      for (int c = 0; c < EED; ++c)
        F[c] += a * X[(((size_t)(b * EED + c)) << 14) + (i << 7) + j];
    }
  }
#pragma unroll
  for (int c = 0; c < EED; ++c) {
    float v;
    if (Wmix) {
      v = 0.0f;
#pragma unroll
      for (int cc = 0; cc < EED; ++cc) v += Wmix[c * EED + cc] * F[cc];
    } else {
      v = F[c];
    }
    E[((size_t)(b * EED + c)) * NN + i] = v;
  }
}

// ---------------------------------------------------------------------------
// Fused node update:
//   ln[co]   = (deg * sum_{node ch in group} Wnb*node + sum_{edge ch} Wnb*E) * rowmax
//   next[co] = (W_node groupconv of node)[co] * nz * ln[co]
// Group structure of the 45-ch concat: groups 0-3 pure node; group 4 mixes
// node ch 36-39 with the 5 edge channels.
// ---------------------------------------------------------------------------
__global__ void k_node(const float* __restrict__ node_cur,
                       const float* __restrict__ deg,
                       const float* __restrict__ rowmax,
                       const float* __restrict__ nz,
                       const float* __restrict__ E,
                       const float* __restrict__ Wnb_r,   // 5*8*9
                       const float* __restrict__ Wn_r,    // 20*2*2
                       float* __restrict__ node_next)
{
  int b = blockIdx.x, i = threadIdx.x;
  float nc[NAL], ee[EED];
#pragma unroll 4
  for (int c = 0; c < NAL; ++c) nc[c] = node_cur[((size_t)(b * NAL + c)) * NN + i];
#pragma unroll
  for (int c = 0; c < EED; ++c) ee[c] = E[((size_t)(b * EED + c)) * NN + i];
  float d = deg[b * NN + i];
  float rm = rowmax[b * NN + i];
  float z = nz[b * NN + i];
  for (int co = 0; co < NAL; ++co) {
    int g = co >> 3, o = co & 7;
    float acc_n = 0.0f, acc_e = 0.0f;
#pragma unroll
    for (int k = 0; k < 9; ++k) {
      int ch = 9 * g + k;
      float w = Wnb_r[(g * 8 + o) * 9 + k];
      if (ch < NAL) acc_n += w * nc[ch];
      else          acc_e += w * ee[ch - NAL];
    }
    float ln = (d * acc_n + acc_e) * rm;
    int gg = co >> 1, oo = co & 1;
    float nnv = Wn_r[(gg * 2 + oo) * 2 + 0] * nc[2 * gg] +
                Wn_r[(gg * 2 + oo) * 2 + 1] * nc[2 * gg + 1];
    node_next[((size_t)(b * NAL + co)) * NN + i] = nnv * z * ln;
  }
}

// ---------------------------------------------------------------------------
// WMMA batched matmul: for each batch b, 6 products sharing A = adj_cur:
//   mat 0..4 : M[b,c]  = adj @ edge_data[b,c]   (col-major f16 B)
//   mat 5    : nxt[b]  = adj @ adj0             (col-major f16 B)
// One wave -> one 16x16 C tile, K = 128 via 4x V_WMMA_F32_16X16X32_F16.
// Fragment layouts (this code's contiguous K order) (lanes 0-15: K 0-15 / M=r; lanes
// 16-31: K 16-31 / M=8+r).
// ---------------------------------------------------------------------------
__global__ void k_mm(const _Float16* __restrict__ A,      // [B][N][N] row-major
                     const _Float16* __restrict__ Ecm,    // [B][5][N][N] col-major
                     const _Float16* __restrict__ A0cm,   // [B][N][N] col-major
                     float* __restrict__ M,               // [B][5][N][N]
                     float* __restrict__ nxt)              // [B][N][N]
{
  const int TILES = 64, NMAT = 6;
  int wave = (blockIdx.x * blockDim.x + threadIdx.x) >> 5;
  int lane = threadIdx.x & 31;
  int tile = wave % TILES;
  int mat  = (wave / TILES) % NMAT;
  int b    = wave / (TILES * NMAT);
  int i0 = (tile >> 3) << 4;
  int j0 = (tile & 7) << 4;

  const _Float16* Abase = A + ((size_t)b << 14);
  const _Float16* Bbase = (mat < 5) ? (Ecm + (((size_t)(b * EED + mat)) << 14))
                                    : (A0cm + ((size_t)b << 14));
  float* Out = (mat < 5) ? (M + (((size_t)(b * EED + mat)) << 14))
                         : (nxt + ((size_t)b << 14));

  int mrow = lane & 15;            // A row / B column within tile
  int kgrp = (lane >> 4) << 4;     // lanes 16-31 carry K+16

  v8f c = {};
#pragma unroll
  for (int kb = 0; kb < NN; kb += 32) {
    const _Float16* ap = Abase + (size_t)(i0 + mrow) * NN + kb + kgrp;
    const _Float16* bp = Bbase + (size_t)(j0 + mrow) * NN + kb + kgrp;
    if (kb + 32 < NN) {
      __builtin_prefetch((const void*)(ap + 32), 0, 1);
      __builtin_prefetch((const void*)(bp + 32), 0, 1);
    }
    v16h af = *(const v16h*)ap;
    v16h bf = *(const v16h*)bp;
    c = __builtin_amdgcn_wmma_f32_16x16x32_f16(
        /*neg_a=*/false, af, /*neg_b=*/false, bf,
        /*c_mod=*/(short)0, c, /*reuse_a=*/false, /*reuse_b=*/false);
  }
  int cm = (lane >> 4) << 3;   // rows r (lanes 0-15) / r+8 (lanes 16-31)
  int cn = lane & 15;
#pragma unroll
  for (int r = 0; r < 8; ++r)
    Out[(size_t)(i0 + cm + r) * NN + (j0 + cn)] = c[r];
}

// ---------------------------------------------------------------------------
// Adjacency update: adj_sum += adj_cur;
// adj_cur = max( (nxt!=0) - adj_sum - N*eye, 0 )  (stays exactly 0/1)
// ---------------------------------------------------------------------------
__global__ void k_adj(const float* __restrict__ nxt,
                      float* __restrict__ adj_sum,
                      float* __restrict__ adj_cur,
                      _Float16* __restrict__ adj_cur_h)
{
  int t = blockIdx.x * blockDim.x + threadIdx.x;
  int rem = t & 16383;
  int i = rem >> 7, j = rem & 127;
  float s = adj_sum[t] + adj_cur[t];
  adj_sum[t] = s;
  float bin = (nxt[t] != 0.0f) ? 1.0f : 0.0f;
  float v = fmaxf(bin - s - (i == j ? (float)NN : 0.0f), 0.0f);
  adj_cur[t] = v;
  adj_cur_h[t] = (_Float16)v;
}

__global__ void k_degrow(const float* __restrict__ adj_cur,
                         float* __restrict__ deg,
                         float* __restrict__ rowmax)
{
  int b = blockIdx.x, i = threadIdx.x;
  const float* row = adj_cur + ((size_t)(b * NN + i)) * NN;
  float d = 0.0f;
  for (int j = 0; j < NN; ++j) d += row[j];
  deg[b * NN + i] = d;
  rowmax[b * NN + i] = (d > 0.0f) ? 1.0f : 0.0f;
}

// ---------------------------------------------------------------------------
// Head: stage mix, BN(eval), Linear.
// ---------------------------------------------------------------------------
__global__ void k_head(const float* __restrict__ fps,     // [3][B][50]
                       const float* __restrict__ Wst,     // [3]
                       const float* __restrict__ bng, const float* __restrict__ bnb,
                       const float* __restrict__ bnm, const float* __restrict__ bnv,
                       const float* __restrict__ Wc, const float* __restrict__ bc,
                       float* __restrict__ out)           // [B][12]
{
  __shared__ float xs[NGC1];
  int b = blockIdx.x, t = threadIdx.x;
  if (t < NGC1) {
    float x = Wst[0] * fps[(0 * BB + b) * NGC1 + t] +
              Wst[1] * fps[(1 * BB + b) * NGC1 + t] +
              Wst[2] * fps[(2 * BB + b) * NGC1 + t];
    x = (x - bnm[t]) * rsqrtf(bnv[t] + 1e-5f) * bng[t] + bnb[t];
    xs[t] = x;
  }
  __syncthreads();
  if (t < NCLASS) {
    float acc = bc[t];
#pragma unroll 10
    for (int f = 0; f < NGC1; ++f) acc += xs[f] * Wc[t * NGC1 + f];
    out[b * NCLASS + t] = acc;
  }
}

// ---------------------------------------------------------------------------
extern "C" void kernel_launch(void* const* d_in, const int* in_sizes, int n_in,
                              void* d_out, int out_size, void* d_ws, size_t ws_size,
                              hipStream_t stream) {
  (void)in_sizes; (void)n_in; (void)out_size; (void)ws_size;

  const float* adjs     = (const float*)d_in[0];
  const int*   afms     = (const int*)  d_in[1];
  const float* axfms    = (const float*)d_in[2];
  const int*   bfts     = (const int*)  d_in[3];
  const float* edge_emb = (const float*)d_in[4];
  const float* node_emb = (const float*)d_in[5];
  const float* W_node   = (const float*)d_in[6];   // [3][20][2][2]
  const float* W_out    = (const float*)d_in[7];   // [3][10][5][4]
  const float* W_nb     = (const float*)d_in[8];   // [3][5][8][9]
  const float* W_edge   = (const float*)d_in[9];   // [3][1][5][5]
  const float* W_stage  = (const float*)d_in[10];
  const float* bn_g     = (const float*)d_in[11];
  const float* bn_b     = (const float*)d_in[12];
  const float* bn_m     = (const float*)d_in[13];
  const float* bn_v     = (const float*)d_in[14];
  const float* Wc       = (const float*)d_in[15];
  const float* bc       = (const float*)d_in[16];
  float* out = (float*)d_out;

  // ---- workspace carve-up (all 256B aligned; total ~77 MB) ----
  size_t off = 0;
  auto alloc = [&](size_t bytes) -> void* {
    void* p = (void*)((char*)d_ws + off);
    off += (bytes + 255) & ~(size_t)255;
    return p;
  };
  const size_t NNsq = (size_t)NN * NN;
  float*    edge_f32  = (float*)   alloc(BB * EED * NNsq * 4);
  _Float16* edge_hcm  = (_Float16*)alloc(BB * EED * NNsq * 2);
  float*    adj0      = (float*)   alloc(BB * NNsq * 4);
  _Float16* adj0_hcm  = (_Float16*)alloc(BB * NNsq * 2);
  float*    adj_cur   = (float*)   alloc(BB * NNsq * 4);
  _Float16* adj_cur_h = (_Float16*)alloc(BB * NNsq * 2);
  float*    adj_sum   = (float*)   alloc(BB * NNsq * 4);
  float*    nxt       = (float*)   alloc(BB * NNsq * 4);
  float*    Mbuf      = (float*)   alloc(BB * EED * NNsq * 4);
  float*    node_a    = (float*)   alloc((size_t)BB * NAL * NN * 4);
  float*    node_b    = (float*)   alloc((size_t)BB * NAL * NN * 4);
  float*    nz        = (float*)   alloc((size_t)BB * NN * 4);
  float*    deg       = (float*)   alloc((size_t)BB * NN * 4);
  float*    rowmax    = (float*)   alloc((size_t)BB * NN * 4);
  float*    Ebuf      = (float*)   alloc((size_t)BB * EED * NN * 4);
  float*    fps       = (float*)   alloc((size_t)3 * BB * NGC1 * 4);

  // ---- setup ----
  k_setup_edge<<<(BB * NNsq) / 256, 256, 0, stream>>>(
      adjs, bfts, edge_emb, edge_f32, edge_hcm, adj0, adj0_hcm,
      adj_cur, adj_cur_h, adj_sum);
  k_setup_node<<<BB, NN, 0, stream>>>(
      adjs, afms, axfms, node_emb, adj0, node_a, nz, deg, rowmax);

  // ---- radius loop ----
  float* ncur = node_a;
  float* nnext = node_b;
  for (int r = 0; r < 3; ++r) {
    k_fp<<<BB, 64, 0, stream>>>(ncur, nz, W_out + r * 200, fps + (size_t)r * BB * NGC1);
    if (r == 2) break;  // node/edge/adj outputs of the last radius are dead

    // E_r: r==0 reduces original edge_data; r>=1 reduces M_{r-1} + W_edge mix
    k_emix<<<BB, NN, 0, stream>>>(
        adj_cur, (r == 0) ? edge_f32 : Mbuf,
        (r == 0) ? (const float*)nullptr : (W_edge + (r - 1) * 25), Ebuf);

    k_node<<<BB, NN, 0, stream>>>(
        ncur, deg, rowmax, nz, Ebuf, W_nb + r * 360, W_node + r * 80, nnext);

    // WMMA: M = adj @ edge_data (5 ch), nxt = adj @ adj0
    k_mm<<<(BB * 6 * 64) / 4, 128, 0, stream>>>(
        adj_cur_h, edge_hcm, adj0_hcm, Mbuf, nxt);

    k_adj<<<(BB * NNsq) / 256, 256, 0, stream>>>(nxt, adj_sum, adj_cur, adj_cur_h);
    k_degrow<<<BB, NN, 0, stream>>>(adj_cur, deg, rowmax);

    float* tmp = ncur; ncur = nnext; nnext = tmp;
  }

  // ---- head ----
  k_head<<<BB, 64, 0, stream>>>(fps, W_stage, bn_g, bn_b, bn_m, bn_v, Wc, bc, out);
}
